// ODELSTM_22857815949631
// MI455X (gfx1250) — matrix operand order repaired and checked by reading, not compiled, not run on hardware
//
#include <hip/hip_runtime.h>
#include <hip/hip_bf16.h>
#include <math.h>

// ---------------------------------------------------------------------------
// ODE-LSTM for MI455X (gfx1250): bf16 WMMA 16x16x32 GEMM pipeline, fp32 accum.
// Global->LDS traffic uses CDNA5 async loads (ASYNCcnt), double-buffered LDS.
// B=4096, OBS=256, H=1024, OUT=256, T=20 (19 RK4 steps).
// ---------------------------------------------------------------------------

typedef __bf16 bf16_t;
typedef __attribute__((ext_vector_type(16))) __bf16 v16bf_t;
typedef __attribute__((ext_vector_type(8)))  float  v8f_t;

#define TILE_M 128
#define TILE_N 128
#define TILE_K 32
#define LDSA_STRIDE 80    // bytes per A-tile row: 64 data + 16 pad (conflict break)
#define LDSB_STRIDE 272   // bytes per B-tile row: 256 data + 16 pad
#define LDSA_BUFSZ  (TILE_M * LDSA_STRIDE)
#define LDSB_BUFSZ  (TILE_K * LDSB_STRIDE)

enum GemmMode {
  MODE_SIGMOID = 0,   // dstF = sigmoid(acc + bias)
  MODE_TANH    = 1,   // dstB = (bf16) tanh(acc + bias)
  MODE_K1      = 2,   // k=acc+b; ksum = k;      dstB = bf16(h + 0.5*dt*k)
  MODE_K2      = 3,   // k=acc+b; ksum += 2k;    dstB = bf16(h + 0.5*dt*k)
  MODE_K3      = 4,   // k=acc+b; ksum += 2k;    dstB = bf16(h + dt*k)
  MODE_K4      = 5,   // k=acc+b; hn=h+dt/6*(ksum+k); dstF=hn; dstB=bf16(hn)
  MODE_OUT     = 6    // dstF = acc + bias
};

union FragBf {
  v16bf_t v;
  uint4   q[2];
};

// CDNA5 async copy: 16B global -> LDS, tracked by ASYNCcnt (no VGPR data path).
// VDST operand is the LDS-relative byte address (low 32 bits of a generic LDS
// pointer per the ISA flat-aperture rules).
__device__ __forceinline__ void async_load_b128(unsigned ldsAddr, const void* gaddr) {
  asm volatile("global_load_async_to_lds_b128 %0, %1, off"
               :
               : "v"(ldsAddr), "v"((unsigned long long)(size_t)gaddr)
               : "memory");
}

__device__ __forceinline__ void wait_async0() {
#if __has_builtin(__builtin_amdgcn_s_wait_asynccnt)
  __builtin_amdgcn_s_wait_asynccnt(0);
#else
  asm volatile("s_wait_asynccnt 0x0" ::: "memory");
#endif
}

template <int MODE>
__global__ __launch_bounds__(256)
void gemm_wmma_kernel(const bf16_t* __restrict__ A, int lda,
                      const bf16_t* __restrict__ Bm, int ldb,
                      const float*  __restrict__ bias,
                      int K, int N,
                      float*  __restrict__ dstF,
                      bf16_t* __restrict__ dstB, int ldDstB,
                      const float* __restrict__ hIn,
                      float*  __restrict__ ksum,
                      const float* __restrict__ tarr, int step)
{
  __shared__ __align__(16) unsigned char smem[2 * LDSA_BUFSZ + 2 * LDSB_BUFSZ];

  const int tid  = threadIdx.x;
  const int lane = tid & 31;
  const int wave = tid >> 5;
  const int wm   = (wave & 3) * 32;   // wave M-offset inside tile (4 waves in M)
  const int wn   = (wave >> 2) * 64;  // wave N-offset inside tile (2 waves in N)

  const int mBlock = blockIdx.y * TILE_M;
  const int nBlock = blockIdx.x * TILE_N;
  const int nk     = K / TILE_K;

  // Per-thread fixed segment assignment (A: 4 segs/row, B: 16 segs/row).
  const int arow0 = tid >> 2,       aseg0 = tid & 3;        // segs 0..255
  const int arow1 = (tid + 256) >> 2, aseg1 = tid & 3;      // segs 256..511
  const int brow0 = tid >> 4,       bseg0 = tid & 15;
  const int brow1 = (tid + 256) >> 4, bseg1 = tid & 15;

  const unsigned ldsBase = (unsigned)(size_t)&smem[0];

  auto issueAsync = [&](int kc, int buf) {
    const unsigned baseA = ldsBase + buf * LDSA_BUFSZ;
    const unsigned baseB = ldsBase + 2 * LDSA_BUFSZ + buf * LDSB_BUFSZ;
    const unsigned char* gA = (const unsigned char*)A + (long)kc * TILE_K * 2;
    const unsigned char* gB = (const unsigned char*)Bm +
                              ((long)kc * TILE_K * ldb + nBlock) * 2;
    async_load_b128(baseA + arow0 * LDSA_STRIDE + aseg0 * 16,
                    gA + (long)(mBlock + arow0) * lda * 2 + aseg0 * 16);
    async_load_b128(baseA + arow1 * LDSA_STRIDE + aseg1 * 16,
                    gA + (long)(mBlock + arow1) * lda * 2 + aseg1 * 16);
    async_load_b128(baseB + brow0 * LDSB_STRIDE + bseg0 * 16,
                    gB + (long)brow0 * ldb * 2 + bseg0 * 16);
    async_load_b128(baseB + brow1 * LDSB_STRIDE + bseg1 * 16,
                    gB + (long)brow1 * ldb * 2 + bseg1 * 16);
  };

  v8f_t acc[2][4];
#pragma unroll
  for (int mi = 0; mi < 2; ++mi)
#pragma unroll
    for (int ni = 0; ni < 4; ++ni)
      acc[mi][ni] = {};

  issueAsync(0, 0);
  wait_async0();
  __syncthreads();

  for (int kc = 0; kc < nk; ++kc) {
    const int cur = kc & 1;
    if (kc + 1 < nk) issueAsync(kc + 1, cur ^ 1);

    const unsigned char* curA = smem + cur * LDSA_BUFSZ;
    const unsigned char* curB = smem + 2 * LDSA_BUFSZ + cur * LDSB_BUFSZ;

    // --- fragment loads from LDS per the CDNA5 16-bit WMMA layouts ----------
    FragBf fa[2], fb[4];
    const int hi = lane >> 4;
#pragma unroll
    for (int mi = 0; mi < 2; ++mi) {
      // A 16x32: lanes 0-15 row M=lane K{0..7,16..23}; lanes 16-31 K{8..15,24..31}
      int row = wm + mi * 16 + (lane & 15);
      const unsigned char* base = curA + row * LDSA_STRIDE + hi * 16;
      fa[mi].q[0] = *(const uint4*)(base);
      fa[mi].q[1] = *(const uint4*)(base + 32);
    }
#pragma unroll
    for (int ni = 0; ni < 4; ++ni) {
      // B 32x16: lanes 0-15 col N=lane K{0..7,16..23}; lanes 16-31 K{8..15,24..31}
      const unsigned char* base = curB + (wn + ni * 16 + (lane & 15)) * 2;
#pragma unroll
      for (int s = 0; s < 16; ++s) {
        const int k = (s < 8) ? (hi * 8 + s) : (16 + hi * 8 + (s - 8));
        fb[ni].v[s] = *(const bf16_t*)(base + k * LDSB_STRIDE);
      }
    }

#pragma unroll
    for (int mi = 0; mi < 2; ++mi)
#pragma unroll
      for (int ni = 0; ni < 4; ++ni)
        acc[mi][ni] = __builtin_amdgcn_wmma_f32_16x16x32_bf16(
            false, fa[mi].v, false, fb[ni].v, (short)0, acc[mi][ni], false, false);

    if (kc + 1 < nk) wait_async0();
    __syncthreads();
  }

  // --- fused epilogue (compile-time specialized) ---------------------------
  float dt = 0.f;
  if constexpr (MODE >= MODE_K1 && MODE <= MODE_K4) dt = tarr[step + 1] - tarr[step];

#pragma unroll
  for (int mi = 0; mi < 2; ++mi) {
#pragma unroll
    for (int ni = 0; ni < 4; ++ni) {
      const int gn = nBlock + wn + ni * 16 + (lane & 15);
      const float bv = bias[gn];
#pragma unroll
      for (int r = 0; r < 8; ++r) {
        // C/D layout: lanes 0-15 col N=lane, VGPR r row M=r; lanes 16-31 row M=r+8
        const int gm = mBlock + wm + mi * 16 + (lane >> 4) * 8 + r;
        const long idx = (long)gm * N + gn;
        const float v = acc[mi][ni][r] + bv;
        if constexpr (MODE == MODE_SIGMOID) {
          dstF[idx] = 1.f / (1.f + __expf(-v));
        } else if constexpr (MODE == MODE_TANH) {
          dstB[(long)gm * ldDstB + gn] = (bf16_t)tanhf(v);
        } else if constexpr (MODE == MODE_K1) {
          float h0 = hIn[idx];
          ksum[idx] = v;
          dstB[(long)gm * ldDstB + gn] = (bf16_t)(h0 + 0.5f * dt * v);
        } else if constexpr (MODE == MODE_K2) {
          float h0 = hIn[idx];
          ksum[idx] += 2.f * v;
          dstB[(long)gm * ldDstB + gn] = (bf16_t)(h0 + 0.5f * dt * v);
        } else if constexpr (MODE == MODE_K3) {
          float h0 = hIn[idx];
          ksum[idx] += 2.f * v;
          dstB[(long)gm * ldDstB + gn] = (bf16_t)(h0 + dt * v);
        } else if constexpr (MODE == MODE_K4) {
          float h0 = hIn[idx];
          float hn = h0 + (dt * (1.f / 6.f)) * (ksum[idx] + v);
          dstF[idx] = hn;
          dstB[(long)gm * ldDstB + gn] = (bf16_t)hn;
        } else {  // MODE_OUT
          dstF[idx] = v;
        }
      }
    }
  }
}

// --------------------------- helper kernels --------------------------------

__global__ void convert_f32_bf16_kernel(const float* __restrict__ src,
                                        bf16_t* __restrict__ dst, long n) {
  long i = (long)blockIdx.x * blockDim.x + threadIdx.x;
  if (i < n) dst[i] = (bf16_t)src[i];
}

// Pack [x | h] -> bf16 xh (4096 x 1280) and copy h -> fp32 ODE state buffer.
__global__ void pack_xh_kernel(const float* __restrict__ x,
                               const float* __restrict__ h,
                               bf16_t* __restrict__ xh,
                               float* __restrict__ hode) {
  long i = (long)blockIdx.x * blockDim.x + threadIdx.x;
  if (i >= 4096L * 1280) return;
  int row = (int)(i / 1280), col = (int)(i % 1280);
  float v;
  if (col < 256) {
    v = x[(long)row * 256 + col];
  } else {
    v = h[(long)row * 1024 + (col - 256)];
    hode[(long)row * 1024 + (col - 256)] = v;
  }
  xh[i] = (bf16_t)v;
}

// c_new = g*(c + c_tilde); h_new = g*tanh(c_new); also bf16 copy of h_new.
__global__ void lstm_elem_kernel(const float* __restrict__ gate,
                                 const float* __restrict__ c,
                                 const float* __restrict__ ctilde,
                                 float* __restrict__ hnewF,
                                 float* __restrict__ cnewF,
                                 bf16_t* __restrict__ hnewB, long n) {
  long i = (long)blockIdx.x * blockDim.x + threadIdx.x;
  if (i >= n) return;
  float g  = gate[i];
  float cn = g * (c[i] + ctilde[i]);
  float hn = g * tanhf(cn);
  cnewF[i] = cn;
  hnewF[i] = hn;
  hnewB[i] = (bf16_t)hn;
}

// ------------------------------- launcher ----------------------------------

extern "C" void kernel_launch(void* const* d_in, const int* in_sizes, int n_in,
                              void* d_out, int out_size, void* d_ws, size_t ws_size,
                              hipStream_t stream) {
  (void)in_sizes; (void)n_in; (void)out_size; (void)ws_size;

  const float* x     = (const float*)d_in[0];
  const float* h     = (const float*)d_in[1];
  const float* c     = (const float*)d_in[2];
  const float* t     = (const float*)d_in[3];
  const float* W_i2h = (const float*)d_in[4];
  const float* b_i2h = (const float*)d_in[5];
  const float* W_h2o = (const float*)d_in[6];
  const float* b_h2o = (const float*)d_in[7];
  const float* Wf1   = (const float*)d_in[8];
  const float* bf1   = (const float*)d_in[9];
  const float* Wf2   = (const float*)d_in[10];
  const float* bf2   = (const float*)d_in[11];

  const long Bn = 4096, OBS = 256, H = 1024, OUT = 256;
  const int  T  = 20;

  // d_out = [out (B x OUT) | h_new (B x H) | c_new (B x H)]
  float* outO = (float*)d_out;
  float* outH = outO + Bn * OUT;
  float* outC = outH + Bn * H;

  // workspace carve-out
  char* ws = (char*)d_ws;
  auto carve = [&](size_t bytes) {
    char* p = ws;
    ws += (bytes + 255) & ~(size_t)255;
    return p;
  };
  bf16_t* wi2h_b = (bf16_t*)carve((OBS + H) * H * sizeof(bf16_t));
  bf16_t* wf1_b  = (bf16_t*)carve(H * H * sizeof(bf16_t));
  bf16_t* wf2_b  = (bf16_t*)carve(H * H * sizeof(bf16_t));
  bf16_t* wh2o_b = (bf16_t*)carve(H * OUT * sizeof(bf16_t));
  bf16_t* xh     = (bf16_t*)carve(Bn * (OBS + H) * sizeof(bf16_t)); // [x | h] bf16
  bf16_t* zb     = (bf16_t*)carve(Bn * H * sizeof(bf16_t));         // tanh layer
  bf16_t* hstage = (bf16_t*)carve(Bn * H * sizeof(bf16_t));         // RK4 stage input
  bf16_t* hnew_b = (bf16_t*)carve(Bn * H * sizeof(bf16_t));
  float*  hode   = (float*)carve(Bn * H * sizeof(float));           // fp32 ODE state
  float*  ksum   = (float*)carve(Bn * H * sizeof(float));
  float*  gate   = (float*)carve(Bn * H * sizeof(float));
  float*  ctilde = (float*)carve(Bn * H * sizeof(float));

  const dim3 blk(256);
  const dim3 grdH((unsigned)(H / TILE_N),   (unsigned)(Bn / TILE_M)); // (8, 32)
  const dim3 grdO((unsigned)(OUT / TILE_N), (unsigned)(Bn / TILE_M)); // (2, 32)

  // 1) weight conversions fp32 -> bf16
  {
    long n1 = (OBS + H) * H, n2 = H * H, n3 = H * OUT;
    convert_f32_bf16_kernel<<<(unsigned)((n1 + 255) / 256), blk, 0, stream>>>(W_i2h, wi2h_b, n1);
    convert_f32_bf16_kernel<<<(unsigned)((n2 + 255) / 256), blk, 0, stream>>>(Wf1, wf1_b, n2);
    convert_f32_bf16_kernel<<<(unsigned)((n2 + 255) / 256), blk, 0, stream>>>(Wf2, wf2_b, n2);
    convert_f32_bf16_kernel<<<(unsigned)((n3 + 255) / 256), blk, 0, stream>>>(W_h2o, wh2o_b, n3);
  }
  // 2) pack [x|h] to bf16 and init fp32 ODE state
  pack_xh_kernel<<<(unsigned)((Bn * (OBS + H) + 255) / 256), blk, 0, stream>>>(x, h, xh, hode);

  // 3) gate = sigmoid([x,h] @ W_i2h + b)  (K = 1280)
  gemm_wmma_kernel<MODE_SIGMOID><<<grdH, blk, 0, stream>>>(
      xh, (int)(OBS + H), wi2h_b, (int)H, b_i2h, (int)(OBS + H), (int)H,
      gate, nullptr, 0, nullptr, nullptr, nullptr, 0);

  // 4) RK4: h lives as fp32 in hode, bf16 in xh[:,256:] (lda 1280)
  for (int s = 0; s < T - 1; ++s) {
    // k1: f(h)
    gemm_wmma_kernel<MODE_TANH><<<grdH, blk, 0, stream>>>(
        xh + OBS, (int)(OBS + H), wf1_b, (int)H, bf1, (int)H, (int)H,
        nullptr, zb, (int)H, nullptr, nullptr, nullptr, 0);
    gemm_wmma_kernel<MODE_K1><<<grdH, blk, 0, stream>>>(
        zb, (int)H, wf2_b, (int)H, bf2, (int)H, (int)H,
        nullptr, hstage, (int)H, hode, ksum, t, s);
    // k2: f(h + 0.5 dt k1)
    gemm_wmma_kernel<MODE_TANH><<<grdH, blk, 0, stream>>>(
        hstage, (int)H, wf1_b, (int)H, bf1, (int)H, (int)H,
        nullptr, zb, (int)H, nullptr, nullptr, nullptr, 0);
    gemm_wmma_kernel<MODE_K2><<<grdH, blk, 0, stream>>>(
        zb, (int)H, wf2_b, (int)H, bf2, (int)H, (int)H,
        nullptr, hstage, (int)H, hode, ksum, t, s);
    // k3: f(h + 0.5 dt k2)
    gemm_wmma_kernel<MODE_TANH><<<grdH, blk, 0, stream>>>(
        hstage, (int)H, wf1_b, (int)H, bf1, (int)H, (int)H,
        nullptr, zb, (int)H, nullptr, nullptr, nullptr, 0);
    gemm_wmma_kernel<MODE_K3><<<grdH, blk, 0, stream>>>(
        zb, (int)H, wf2_b, (int)H, bf2, (int)H, (int)H,
        nullptr, hstage, (int)H, hode, ksum, t, s);
    // k4: f(h + dt k3); h <- h + dt/6 (k1+2k2+2k3+k4), bf16 copy into xh[:,256:]
    gemm_wmma_kernel<MODE_TANH><<<grdH, blk, 0, stream>>>(
        hstage, (int)H, wf1_b, (int)H, bf1, (int)H, (int)H,
        nullptr, zb, (int)H, nullptr, nullptr, nullptr, 0);
    gemm_wmma_kernel<MODE_K4><<<grdH, blk, 0, stream>>>(
        zb, (int)H, wf2_b, (int)H, bf2, (int)H, (int)H,
        hode, xh + OBS, (int)(OBS + H), hode, ksum, t, s);
  }

  // 5) c_tilde = sigmoid([x, h_ode] @ W_i2h + b)  (xh now holds [x | h_ode])
  gemm_wmma_kernel<MODE_SIGMOID><<<grdH, blk, 0, stream>>>(
      xh, (int)(OBS + H), wi2h_b, (int)H, b_i2h, (int)(OBS + H), (int)H,
      ctilde, nullptr, 0, nullptr, nullptr, nullptr, 0);

  // 6) LSTM cell elementwise -> h_new (fp32 + bf16), c_new
  lstm_elem_kernel<<<(unsigned)((Bn * H + 255) / 256), blk, 0, stream>>>(
      gate, c, ctilde, outH, outC, hnew_b, Bn * H);

  // 7) out = h_new @ W_h2o + b_h2o
  gemm_wmma_kernel<MODE_OUT><<<grdO, blk, 0, stream>>>(
      hnew_b, (int)H, wh2o_b, (int)OUT, b_h2o, (int)H, (int)OUT,
      outO, nullptr, 0, nullptr, nullptr, nullptr, 0);
}
